// HellingerDistance_18305150615591
// MI455X (gfx1250) — matrix operand order repaired and checked by reading, not compiled, not run on hardware
//
#include <hip/hip_runtime.h>

typedef __attribute__((ext_vector_type(16))) _Float16 v16h;
typedef __attribute__((ext_vector_type(8)))  _Float16 v8h;
typedef __attribute__((ext_vector_type(4)))  _Float16 v4h;
typedef __attribute__((ext_vector_type(8)))  float    v8f;

#define TILE 128
#define KC   32
#define LDA  40    // halves per A row in LDS (32 + 8 pad)
#define LDB  136   // halves per B^T row in LDS (128 + 8 pad, 272B = 17*16B)
#define TP   80    // transpose tile pitch (64 + 16 pad, 160B = 10*16B)

// ---------------------------------------------------------------------------
// Preprocess 1: sa16[row][d] = (f16)sqrt(a[row][d]);  rsA[row] = sum_d a[row][d]
// one 256-thread block per row, D = 1024 -> one float4 per thread
// ---------------------------------------------------------------------------
__global__ __launch_bounds__(256)
void conv_rowsum_a(const float* __restrict__ a, _Float16* __restrict__ sa,
                   float* __restrict__ rs, int D)
{
    __shared__ float red[256];
    const int row = blockIdx.x;
    const int t   = threadIdx.x;
    const float4 v = ((const float4*)(a + (size_t)row * D))[t];
    v4h h;
    h[0] = (_Float16)sqrtf(v.x);
    h[1] = (_Float16)sqrtf(v.y);
    h[2] = (_Float16)sqrtf(v.z);
    h[3] = (_Float16)sqrtf(v.w);
    *(v4h*)(sa + (size_t)row * D + 4 * t) = h;
    red[t] = (v.x + v.y) + (v.z + v.w);
    __syncthreads();
    #pragma unroll
    for (int s = 128; s > 0; s >>= 1) {
        if (t < s) red[t] += red[t + s];
        __syncthreads();
    }
    if (t == 0) rs[row] = red[0];
}

// ---------------------------------------------------------------------------
// Preprocess 2: rsB[row] = sum_d b[row][d]
// ---------------------------------------------------------------------------
__global__ __launch_bounds__(256)
void rowsum_b(const float* __restrict__ b, float* __restrict__ rs, int D)
{
    __shared__ float red[256];
    const int row = blockIdx.x;
    const int t   = threadIdx.x;
    const float4 v = ((const float4*)(b + (size_t)row * D))[t];
    red[t] = (v.x + v.y) + (v.z + v.w);
    __syncthreads();
    #pragma unroll
    for (int s = 128; s > 0; s >>= 1) {
        if (t < s) red[t] += red[t + s];
        __syncthreads();
    }
    if (t == 0) rs[row] = red[0];
}

// ---------------------------------------------------------------------------
// Preprocess 3: sbT[d][m] = (f16)sqrt(b[m][d])   (64x64 tiles via LDS)
// ---------------------------------------------------------------------------
__global__ __launch_bounds__(256)
void trans_conv_b(const float* __restrict__ b, _Float16* __restrict__ sbT,
                  int D, int M)
{
    __shared__ _Float16 tile[64 * TP];   // [d][m]
    const int m0 = blockIdx.x * 64;
    const int d0 = blockIdx.y * 64;
    const int t  = threadIdx.x;
    {
        const int mr = t >> 2;           // 0..63
        const int dc = (t & 3) * 16;     // 0,16,32,48
        const float4* src = (const float4*)(b + (size_t)(m0 + mr) * D + d0 + dc);
        #pragma unroll
        for (int j = 0; j < 4; ++j) {
            float4 v = src[j];
            tile[(dc + 4*j + 0) * TP + mr] = (_Float16)sqrtf(v.x);
            tile[(dc + 4*j + 1) * TP + mr] = (_Float16)sqrtf(v.y);
            tile[(dc + 4*j + 2) * TP + mr] = (_Float16)sqrtf(v.z);
            tile[(dc + 4*j + 3) * TP + mr] = (_Float16)sqrtf(v.w);
        }
    }
    __syncthreads();
    {
        const int dr = t >> 2;           // 0..63
        const int mc = (t & 3) * 16;     // 0,16,32,48
        _Float16* dst = sbT + (size_t)(d0 + dr) * M + m0 + mc;
        ((v8h*)dst)[0] = *(v8h*)&tile[dr * TP + mc];
        ((v8h*)dst)[1] = *(v8h*)&tile[dr * TP + mc + 8];
    }
}

// ---------------------------------------------------------------------------
// Main GEMM (fast path): f16 inputs already sqrt'ed; B pre-transposed [D][M].
// 128x128 tile / block, 8 waves (4x2), double-buffered LDS, 1 barrier/chunk.
// ---------------------------------------------------------------------------
__global__ __launch_bounds__(256)
void hell_gemm_f16(const _Float16* __restrict__ sa,
                   const _Float16* __restrict__ sbT,
                   const float* __restrict__ rsa,
                   const float* __restrict__ rsb,
                   float* __restrict__ out, int N, int M, int D)
{
    __shared__ _Float16 sA[2][TILE * LDA];   // [row][k]
    __shared__ _Float16 sB[2][KC * LDB];     // [k][col]
    __shared__ float rsA_s[TILE];
    __shared__ float rsB_s[TILE];

    const int tid  = threadIdx.x;
    const int lane = tid & 31;
    const int wid  = tid >> 5;
    const int waveM = wid & 3;
    const int waveN = wid >> 2;

    const int rowBase = blockIdx.y * TILE;
    const int colBase = blockIdx.x * TILE;

    if (tid < TILE) rsA_s[tid]        = rsa[rowBase + tid];
    else            rsB_s[tid - TILE] = rsb[colBase + tid - TILE];

    // staging assignments: 16 halves (two b128) per thread per matrix
    const int ra_row = tid >> 1;              // 0..127
    const int ra_ks  = (tid & 1) * 16;        // 0 / 16
    const int rb_k   = tid >> 3;              // 0..31
    const int rb_cs  = (tid & 7) * 16;        // 0..112

    const _Float16* gA = sa  + (size_t)(rowBase + ra_row) * D + ra_ks;
    const _Float16* gB = sbT + (size_t)rb_k * M + colBase + rb_cs;

    v8h a0, a1, b0, b1;
    auto LOAD = [&](int k0) {
        const v8h* pa = (const v8h*)(gA + k0);
        a0 = pa[0]; a1 = pa[1];
        const v8h* pb = (const v8h*)(gB + (size_t)k0 * M);
        b0 = pb[0]; b1 = pb[1];
    };
    auto STORE = [&](int buf) {
        v8h* da = (v8h*)&sA[buf][ra_row * LDA + ra_ks];
        da[0] = a0; da[1] = a1;
        v8h* db = (v8h*)&sB[buf][rb_k * LDB + rb_cs];
        db[0] = b0; db[1] = b1;
    };

    v8f acc[2][4] = {};
    union V16 { v16h v; v8h h[2]; };
    const int mrow = lane & 15;
    const int hi   = lane >> 4;

    LOAD(0);
    STORE(0);
    __syncthreads();

    const int niter = D / KC;
    for (int i = 0; i < niter; ++i) {
        if (i + 1 < niter) LOAD((i + 1) * KC);   // overlap with WMMAs below

        const int buf = i & 1;
        // A fragments (ISA 16x32 f16 layout: contiguous 8-half K runs)
        V16 afrag[2];
        #pragma unroll
        for (int tm = 0; tm < 2; ++tm) {
            const _Float16* base = &sA[buf][(waveM*32 + tm*16 + mrow) * LDA];
            afrag[tm].h[0] = *(const v8h*)&base[hi * 8];
            afrag[tm].h[1] = *(const v8h*)&base[16 + hi * 8];
        }
        // B fragments (lane -> N column, K in same table order as A)
        V16 bfrag[4];
        #pragma unroll
        for (int tn = 0; tn < 4; ++tn) {
            const _Float16* base = &sB[buf][waveN*64 + tn*16 + mrow];
            #pragma unroll
            for (int s = 0; s < 8; ++s) {
                bfrag[tn].v[s]     = base[(hi * 8 + s) * LDB];
                bfrag[tn].v[8 + s] = base[(16 + hi * 8 + s) * LDB];
            }
        }
        #pragma unroll
        for (int tm = 0; tm < 2; ++tm)
            #pragma unroll
            for (int tn = 0; tn < 4; ++tn)
                acc[tm][tn] = __builtin_amdgcn_wmma_f32_16x16x32_f16(
                    false, afrag[tm].v, false, bfrag[tn].v,
                    (short)0, acc[tm][tn], false, false);

        if (i + 1 < niter) STORE((i + 1) & 1);
        __syncthreads();
    }

    // epilogue: out = 0.5*(rsA + rsB) - cross
    const int nlo = lane & 15;
    const int mhi = (lane >> 4) * 8;
    #pragma unroll
    for (int tm = 0; tm < 2; ++tm) {
        #pragma unroll
        for (int tn = 0; tn < 4; ++tn) {
            #pragma unroll
            for (int rr = 0; rr < 8; ++rr) {
                const int lm = waveM*32 + tm*16 + rr + mhi;
                const int ln = waveN*64 + tn*16 + nlo;
                const float val = 0.5f * (rsA_s[lm] + rsB_s[ln]) - acc[tm][tn][rr];
                out[(size_t)(rowBase + lm) * M + (colBase + ln)] = val;
            }
        }
    }
}

// ---------------------------------------------------------------------------
// Fallback (no workspace): fused sqrt+convert+rowsum+GEMM (round-1 kernel)
// ---------------------------------------------------------------------------
__global__ __launch_bounds__(256)
void hellinger_fused_kernel(const float* __restrict__ a,
                            const float* __restrict__ b,
                            float* __restrict__ out,
                            int N, int M, int D)
{
    __shared__ _Float16 sA [TILE * LDA];
    __shared__ _Float16 sBT[KC  * LDB];
    __shared__ float    rsA[TILE];
    __shared__ float    rsB[TILE];

    const int tid  = threadIdx.x;
    const int lane = tid & 31;
    const int wid  = tid >> 5;
    const int waveM = wid & 3;
    const int waveN = wid >> 2;

    const int rowBase = blockIdx.y * TILE;
    const int colBase = blockIdx.x * TILE;

    {
        const float* src = (tid < TILE)
            ? (a + (size_t)(rowBase + tid) * D)
            : (b + (size_t)(colBase + (tid - TILE)) * D);
        float s = 0.f;
        const float4* p = (const float4*)src;
        for (int i = 0; i < D / 4; ++i) {
            float4 v = p[i];
            s += (v.x + v.y) + (v.z + v.w);
        }
        if (tid < TILE) rsA[tid] = s;
        else            rsB[tid - TILE] = s;
    }

    v8f acc[2][4] = {};
    const int r  = tid >> 1;
    const int ks = (tid & 1) * 16;
    union V16 { v16h v; v8h h[2]; };

    for (int k0 = 0; k0 < D; k0 += KC) {
        __syncthreads();
        {
            const float4* pa = (const float4*)(a + (size_t)(rowBase + r) * D + k0 + ks);
            _Float16 h[16];
            #pragma unroll
            for (int j = 0; j < 4; ++j) {
                float4 v = pa[j];
                h[4*j+0] = (_Float16)sqrtf(v.x);
                h[4*j+1] = (_Float16)sqrtf(v.y);
                h[4*j+2] = (_Float16)sqrtf(v.z);
                h[4*j+3] = (_Float16)sqrtf(v.w);
            }
            v8h* dst = (v8h*)&sA[r * LDA + ks];
            dst[0] = *(v8h*)&h[0];
            dst[1] = *(v8h*)&h[8];
        }
        {
            const float4* pb = (const float4*)(b + (size_t)(colBase + r) * D + k0 + ks);
            #pragma unroll
            for (int j = 0; j < 4; ++j) {
                float4 v = pb[j];
                sBT[(ks + 4*j + 0) * LDB + r] = (_Float16)sqrtf(v.x);
                sBT[(ks + 4*j + 1) * LDB + r] = (_Float16)sqrtf(v.y);
                sBT[(ks + 4*j + 2) * LDB + r] = (_Float16)sqrtf(v.z);
                sBT[(ks + 4*j + 3) * LDB + r] = (_Float16)sqrtf(v.w);
            }
        }
        __syncthreads();

        const int mrow = lane & 15;
        const int hi   = lane >> 4;
        V16 afrag[2];
        #pragma unroll
        for (int tm = 0; tm < 2; ++tm) {
            const _Float16* base = &sA[(waveM*32 + tm*16 + mrow) * LDA];
            afrag[tm].h[0] = *(const v8h*)&base[hi * 8];
            afrag[tm].h[1] = *(const v8h*)&base[16 + hi * 8];
        }
        V16 bfrag[4];
        #pragma unroll
        for (int tn = 0; tn < 4; ++tn) {
            const _Float16* base = &sBT[waveN*64 + tn*16 + mrow];
            #pragma unroll
            for (int s = 0; s < 8; ++s) {
                bfrag[tn].v[s]     = base[(hi * 8 + s) * LDB];
                bfrag[tn].v[8 + s] = base[(16 + hi * 8 + s) * LDB];
            }
        }
        #pragma unroll
        for (int tm = 0; tm < 2; ++tm)
            #pragma unroll
            for (int tn = 0; tn < 4; ++tn)
                acc[tm][tn] = __builtin_amdgcn_wmma_f32_16x16x32_f16(
                    false, afrag[tm].v, false, bfrag[tn].v,
                    (short)0, acc[tm][tn], false, false);
    }

    __syncthreads();
    const int nlo = lane & 15;
    const int mhi = (lane >> 4) * 8;
    #pragma unroll
    for (int tm = 0; tm < 2; ++tm) {
        #pragma unroll
        for (int tn = 0; tn < 4; ++tn) {
            #pragma unroll
            for (int rr = 0; rr < 8; ++rr) {
                const int lm = waveM*32 + tm*16 + rr + mhi;
                const int ln = waveN*64 + tn*16 + nlo;
                const float val = 0.5f * (rsA[lm] + rsB[ln]) - acc[tm][tn][rr];
                out[(size_t)(rowBase + lm) * M + (colBase + ln)] = val;
            }
        }
    }
}

// ---------------------------------------------------------------------------
extern "C" void kernel_launch(void* const* d_in, const int* in_sizes, int n_in,
                              void* d_out, int out_size, void* d_ws, size_t ws_size,
                              hipStream_t stream)
{
    const float* a = (const float*)d_in[0];
    const float* b = (const float*)d_in[1];
    float* out = (float*)d_out;

    const int D = 1024;                 // fixed by the reference problem
    const int N = in_sizes[0] / D;      // 8192
    const int M = in_sizes[1] / D;      // 8192

    // workspace layout (256B aligned regions)
    const size_t szA  = (size_t)N * D * sizeof(_Float16);
    const size_t offB = (szA + 255) & ~(size_t)255;
    const size_t szB  = (size_t)D * M * sizeof(_Float16);
    const size_t offRA = (offB + szB + 255) & ~(size_t)255;
    const size_t offRB = (offRA + (size_t)N * sizeof(float) + 255) & ~(size_t)255;
    const size_t need  = offRB + (size_t)M * sizeof(float);

    dim3 gemm_grid(M / TILE, N / TILE);

    if (d_ws != nullptr && ws_size >= need) {
        _Float16* sa16 = (_Float16*)d_ws;
        _Float16* sbT  = (_Float16*)((char*)d_ws + offB);
        float*    rsA  = (float*)((char*)d_ws + offRA);
        float*    rsB  = (float*)((char*)d_ws + offRB);

        conv_rowsum_a<<<dim3(N), dim3(256), 0, stream>>>(a, sa16, rsA, D);
        rowsum_b     <<<dim3(M), dim3(256), 0, stream>>>(b, rsB, D);
        trans_conv_b <<<dim3(M / 64, D / 64), dim3(256), 0, stream>>>(b, sbT, D, M);
        hell_gemm_f16<<<gemm_grid, dim3(256), 0, stream>>>(sa16, sbT, rsA, rsB,
                                                           out, N, M, D);
    } else {
        hellinger_fused_kernel<<<gemm_grid, dim3(256), 0, stream>>>(a, b, out, N, M, D);
    }
}
